// DeepseekMoE_16587163697435
// MI455X (gfx1250) — compile-verified, matrix operand order checked by reading, not run on hardware
//
#include <hip/hip_runtime.h>
#include <hip/hip_bf16.h>

typedef __bf16 bf16;
typedef __attribute__((ext_vector_type(16))) __bf16 v16bf;
typedef __attribute__((ext_vector_type(8)))  float  v8f;

#define TOPK 6

union FragU { v16bf v; uint4 q[2]; };

// Load a 16-element bf16 WMMA fragment for lane `lane`.
// Layout (ISA 7.12.2, 16-bit A 16x32): lane r=lane&15 -> row, hi=lane>>4 selects
// K-halves: elements are K = hi*8 + 0..7 (vgpr0-3) and K = 16 + hi*8 + 0..7 (vgpr4-7).
// `p` must already point at  base + row*ld + k0 + hi*8.
__device__ __forceinline__ v16bf ldfrag(const bf16* p) {
    FragU f;
    f.q[0] = *(const uint4*)(p);
    f.q[1] = *(const uint4*)(p + 16);
    return f.v;
}

__device__ __forceinline__ v8f wmma_bf16(v16bf a, v16bf b, v8f c) {
    return __builtin_amdgcn_wmma_f32_16x16x32_bf16(false, a, false, b, (short)0, c, false, false);
}

// ---------------- fp32 -> bf16 conversion (vectorized: float4 in, b64 out) ----------------
__global__ void cvt_bf16_kernel(const float4* __restrict__ src, ushort4* __restrict__ dst, size_t n4) {
    size_t i = (size_t)blockIdx.x * blockDim.x + threadIdx.x;
    size_t stride = (size_t)gridDim.x * blockDim.x;
    for (; i < n4; i += stride) {
        float4 f = src[i];
        union { bf16 b[4]; ushort4 u; } c;
        c.b[0] = (bf16)f.x; c.b[1] = (bf16)f.y; c.b[2] = (bf16)f.z; c.b[3] = (bf16)f.w;
        dst[i] = c.u;
    }
}

// ---------------- router: softmax + top-6 + renorm ----------------
__global__ void router_kernel(const float* __restrict__ x, const float* __restrict__ gw,
                              int* __restrict__ topk_idx, float* __restrict__ topk_w,
                              int D) {
    int n = blockIdx.x;
    int e = threadIdx.x;          // 64 threads, one expert each
    __shared__ float sl[64];
    const float4* xr = (const float4*)(x + (size_t)n * D);
    const float4* wr = (const float4*)(gw + (size_t)e * D);
    float acc = 0.f;
    for (int i = 0; i < D / 4; ++i) {
        float4 a = xr[i], b = wr[i];
        acc += a.x * b.x + a.y * b.y + a.z * b.z + a.w * b.w;
    }
    sl[e] = acc;
    __syncthreads();
    if (e == 0) {
        float mx = -1e30f;
        for (int i = 0; i < 64; ++i) mx = fmaxf(mx, sl[i]);
        float p[64]; float sum = 0.f;
        for (int i = 0; i < 64; ++i) { p[i] = __expf(sl[i] - mx); sum += p[i]; }
        float inv = 1.f / sum;
        int sel[TOPK]; float selw[TOPK]; float wsum = 0.f;
        for (int k = 0; k < TOPK; ++k) {
            float best = -1.f; int bi = 0;
            for (int i = 0; i < 64; ++i) if (p[i] > best) { best = p[i]; bi = i; }
            sel[k] = bi; selw[k] = best * inv; wsum += best * inv; p[bi] = -2.f;
        }
        float r = 1.f / (wsum + 1e-20f);
        for (int k = 0; k < TOPK; ++k) {
            topk_idx[n * TOPK + k] = sel[k];
            topk_w[n * TOPK + k]   = selw[k] * r;
        }
    }
}

// ---------------- slot assignment ----------------
__global__ void assign_kernel(const int* __restrict__ topk_idx, const float* __restrict__ topk_w,
                              int* __restrict__ cnt, int* __restrict__ slot_of_assign,
                              int* __restrict__ token_of_slot, float* __restrict__ weight_of_slot,
                              int A, int C) {
    int a = blockIdx.x * blockDim.x + threadIdx.x;
    if (a >= A) return;
    int e = topk_idx[a];
    int slot = atomicAdd(&cnt[e], 1);
    if (slot < C) {
        int s = e * C + slot;
        slot_of_assign[a] = s;
        token_of_slot[s] = a / TOPK;
        weight_of_slot[s] = topk_w[a];
    } else {
        slot_of_assign[a] = -1;
    }
}

// ---------------- dispatch: copy token rows into [E,C,D] bf16 buffer ----------------
__global__ void dispatch_kernel(const bf16* __restrict__ xb, const int* __restrict__ slot_of_assign,
                                bf16* __restrict__ buf, int D) {
    int a = blockIdx.x;
    int s = slot_of_assign[a];
    if (s < 0) return;
    int tok = a / TOPK;
    const uint4* src = (const uint4*)(xb + (size_t)tok * D);
    uint4* dst = (uint4*)(buf + (size_t)s * D);
    dst[threadIdx.x] = src[threadIdx.x];   // 256 threads x 16B = 4KB = full row
}

// ---------------- SwiGLU GEMM: H = silu(A @ Bg^T) * (A @ Bu^T), bf16 out ----------------
// A:[M,K] Bg,Bu:[Ncol,K] all row-major bf16 (TN).  blockIdx.y = expert.
// Each wave computes a 32x32 output super-tile (2x2 WMMA tiles, gate+up fused):
// per K-step 6 fragment loads feed 8 v_wmma.
__global__ __launch_bounds__(256)
void gemm_swiglu_kernel(const bf16* __restrict__ A, const bf16* __restrict__ Bg,
                        const bf16* __restrict__ Bu, bf16* __restrict__ H,
                        int M, int Ncol, int K,
                        long strideA, long strideB, long strideH) {
    int lane = threadIdx.x & 31;
    int wave = threadIdx.x >> 5;
    int tilesN = Ncol >> 5;                 // 32-wide super-tiles
    int tilesM = M >> 5;
    int tile = blockIdx.x * 8 + wave;
    if (tile >= tilesM * tilesN) return;
    int tm = tile / tilesN, tn = tile % tilesN;
    int e = blockIdx.y;
    int r = lane & 15, hi = lane >> 4;

    const bf16* a0 = A  + (size_t)e * strideA + (size_t)(tm * 32 + r) * K + hi * 8;
    const bf16* a1 = a0 + (size_t)16 * K;
    const bf16* g0 = Bg + (size_t)e * strideB + (size_t)(tn * 32 + r) * K + hi * 8;
    const bf16* g1 = g0 + (size_t)16 * K;
    const bf16* u0 = Bu + (size_t)e * strideB + (size_t)(tn * 32 + r) * K + hi * 8;
    const bf16* u1 = u0 + (size_t)16 * K;

    v8f accg[2][2] = {};
    v8f accu[2][2] = {};
    for (int k = 0; k < K; k += 32) {
        __builtin_prefetch(a0 + k + 256, 0, 1);
        __builtin_prefetch(g0 + k + 256, 0, 1);
        __builtin_prefetch(u0 + k + 256, 0, 1);
        v16bf a[2]  = { ldfrag(a0 + k), ldfrag(a1 + k) };
        v16bf bg[2] = { ldfrag(g0 + k), ldfrag(g1 + k) };
        v16bf bu[2] = { ldfrag(u0 + k), ldfrag(u1 + k) };
#pragma unroll
        for (int i = 0; i < 2; ++i)
#pragma unroll
            for (int j = 0; j < 2; ++j) {
                accg[i][j] = wmma_bf16(a[i], bg[j], accg[i][j]);
                accu[i][j] = wmma_bf16(a[i], bu[j], accu[i][j]);
            }
    }
    bf16* hrow = H + (size_t)e * strideH;
#pragma unroll
    for (int i = 0; i < 2; ++i)
#pragma unroll
        for (int j = 0; j < 2; ++j) {
            int n = tn * 32 + j * 16 + r;
#pragma unroll
            for (int v = 0; v < 8; ++v) {
                int m = tm * 32 + i * 16 + v + hi * 8;
                float g = accg[i][j][v], u = accu[i][j][v];
                float s = (g / (1.f + __expf(-g))) * u;
                hrow[(size_t)m * Ncol + n] = (bf16)s;
            }
        }
}

// ---------------- routed down GEMM + weighted atomic scatter ----------------
// H:[E,C,F] bf16, Wd:[E,D,F] bf16 (row d contiguous over f -> TN). out += w * (H @ Wd^T)
// Each wave computes a 32x64 super-tile (2x4 WMMA tiles): 6 fragment loads -> 8 v_wmma / K-step.
__global__ __launch_bounds__(256)
void gemm_down_scatter_kernel(const bf16* __restrict__ H, const bf16* __restrict__ Wd,
                              const int* __restrict__ token_of_slot,
                              const float* __restrict__ weight_of_slot,
                              float* __restrict__ y, int C, int D, int F) {
    int lane = threadIdx.x & 31;
    int wave = threadIdx.x >> 5;
    int tilesN = D >> 6;                    // 64-wide super-tiles
    int tilesM = C >> 5;
    int tile = blockIdx.x * 8 + wave;
    if (tile >= tilesM * tilesN) return;
    int tm = tile / tilesN, tn = tile % tilesN;
    int e = blockIdx.y;
    int r = lane & 15, hi = lane >> 4;

    const bf16* a0 = H  + (size_t)e * C * F + (size_t)(tm * 32 + r) * F + hi * 8;
    const bf16* a1 = a0 + (size_t)16 * F;
    const bf16* b0 = Wd + (size_t)e * D * F + (size_t)(tn * 64 + r) * F + hi * 8;

    v8f acc[2][4] = {};
    for (int k = 0; k < F; k += 32) {
        __builtin_prefetch(a0 + k + 256, 0, 1);
        __builtin_prefetch(b0 + k + 256, 0, 1);
        v16bf a[2] = { ldfrag(a0 + k), ldfrag(a1 + k) };
        v16bf b[4];
#pragma unroll
        for (int j = 0; j < 4; ++j) b[j] = ldfrag(b0 + (size_t)j * 16 * F + k);
#pragma unroll
        for (int i = 0; i < 2; ++i)
#pragma unroll
            for (int j = 0; j < 4; ++j)
                acc[i][j] = wmma_bf16(a[i], b[j], acc[i][j]);
    }
#pragma unroll
    for (int i = 0; i < 2; ++i)
#pragma unroll
        for (int v = 0; v < 8; ++v) {
            int m = tm * 32 + i * 16 + v + hi * 8;
            int s = e * C + m;
            int tok = token_of_slot[s];
            if (tok >= 0) {
                float w = weight_of_slot[s];
#pragma unroll
                for (int j = 0; j < 4; ++j) {
                    int n = tn * 64 + j * 16 + r;
                    atomicAdd(&y[(size_t)tok * D + n], acc[i][j][v] * w);
                }
            }
        }
}

// ---------------- shared-expert down GEMM: y += H @ Wd^T (exclusive tiles) ----------------
__global__ __launch_bounds__(256)
void gemm_down_add_kernel(const bf16* __restrict__ H, const bf16* __restrict__ Wd,
                          float* __restrict__ y, int M, int D, int F) {
    int lane = threadIdx.x & 31;
    int wave = threadIdx.x >> 5;
    int tilesN = D >> 6;
    int tilesM = M >> 5;
    int tile = blockIdx.x * 8 + wave;
    if (tile >= tilesM * tilesN) return;
    int tm = tile / tilesN, tn = tile % tilesN;
    int r = lane & 15, hi = lane >> 4;

    const bf16* a0 = H  + (size_t)(tm * 32 + r) * F + hi * 8;
    const bf16* a1 = a0 + (size_t)16 * F;
    const bf16* b0 = Wd + (size_t)(tn * 64 + r) * F + hi * 8;

    v8f acc[2][4] = {};
    for (int k = 0; k < F; k += 32) {
        __builtin_prefetch(a0 + k + 256, 0, 1);
        __builtin_prefetch(b0 + k + 256, 0, 1);
        v16bf a[2] = { ldfrag(a0 + k), ldfrag(a1 + k) };
        v16bf b[4];
#pragma unroll
        for (int j = 0; j < 4; ++j) b[j] = ldfrag(b0 + (size_t)j * 16 * F + k);
#pragma unroll
        for (int i = 0; i < 2; ++i)
#pragma unroll
            for (int j = 0; j < 4; ++j)
                acc[i][j] = wmma_bf16(a[i], b[j], acc[i][j]);
    }
#pragma unroll
    for (int i = 0; i < 2; ++i)
#pragma unroll
        for (int v = 0; v < 8; ++v) {
            int m = tm * 32 + i * 16 + v + hi * 8;
#pragma unroll
            for (int j = 0; j < 4; ++j) {
                int n = tn * 64 + j * 16 + r;
                y[(size_t)m * D + n] += acc[i][j][v];
            }
        }
}

extern "C" void kernel_launch(void* const* d_in, const int* in_sizes, int n_in,
                              void* d_out, int out_size, void* d_ws, size_t ws_size,
                              hipStream_t stream) {
    (void)in_sizes; (void)n_in; (void)ws_size;
    constexpr int Bb = 4, S = 2048, D = 2048, E = 64, F = 1408, FS = 2816;
    const int N = Bb * S;                          // 8192 tokens
    const int A = N * TOPK;                        // 49152 assignments
    const int C = 2 * ((A + E - 1) / E);           // 1536 capacity

    const float* x      = (const float*)d_in[0];
    const float* gate_w = (const float*)d_in[1];
    const float* w_gate = (const float*)d_in[2];
    const float* w_up   = (const float*)d_in[3];
    const float* w_down = (const float*)d_in[4];
    const float* sh_g   = (const float*)d_in[5];
    const float* sh_u   = (const float*)d_in[6];
    const float* sh_d   = (const float*)d_in[7];
    float* y = (float*)d_out;

    // -------- workspace carve --------
    char* wsp = (char*)d_ws;
    auto alloc = [&](size_t bytes) -> void* {
        void* p = wsp; wsp += (bytes + 255) & ~(size_t)255; return p;
    };
    bf16* xb   = (bf16*)alloc((size_t)N * D * 2);
    bf16* wgb  = (bf16*)alloc((size_t)E * F * D * 2);
    bf16* wub  = (bf16*)alloc((size_t)E * F * D * 2);
    bf16* wdb  = (bf16*)alloc((size_t)E * D * F * 2);
    bf16* sgb  = (bf16*)alloc((size_t)FS * D * 2);
    bf16* subb = (bf16*)alloc((size_t)FS * D * 2);
    bf16* sdb  = (bf16*)alloc((size_t)D * FS * 2);
    bf16* buf  = (bf16*)alloc((size_t)E * C * D * 2);
    bf16* hbuf = (bf16*)alloc((size_t)E * C * F * 2);
    bf16* hs   = (bf16*)alloc((size_t)N * FS * 2);
    int*   topk_idx       = (int*)alloc((size_t)A * 4);
    float* topk_w         = (float*)alloc((size_t)A * 4);
    int*   cnt            = (int*)alloc((size_t)E * 4);
    int*   slot_of_assign = (int*)alloc((size_t)A * 4);
    int*   token_of_slot  = (int*)alloc((size_t)E * C * 4);
    float* weight_of_slot = (float*)alloc((size_t)E * C * 4);

    // -------- per-call init (graph-capture safe) --------
    hipMemsetAsync(y, 0, (size_t)out_size * sizeof(float), stream);
    hipMemsetAsync(cnt, 0, (size_t)E * 4, stream);
    hipMemsetAsync(token_of_slot, 0xFF, (size_t)E * C * 4, stream);   // -1 = empty slot
    hipMemsetAsync(buf, 0, (size_t)E * C * D * 2, stream);            // zero rows for dropped slots

    // -------- fp32 -> bf16 conversions --------
    cvt_bf16_kernel<<<1024, 256, 0, stream>>>((const float4*)x,      (ushort4*)xb,   (size_t)N * D / 4);
    cvt_bf16_kernel<<<2048, 256, 0, stream>>>((const float4*)w_gate, (ushort4*)wgb,  (size_t)E * F * D / 4);
    cvt_bf16_kernel<<<2048, 256, 0, stream>>>((const float4*)w_up,   (ushort4*)wub,  (size_t)E * F * D / 4);
    cvt_bf16_kernel<<<2048, 256, 0, stream>>>((const float4*)w_down, (ushort4*)wdb,  (size_t)E * D * F / 4);
    cvt_bf16_kernel<<<512,  256, 0, stream>>>((const float4*)sh_g,   (ushort4*)sgb,  (size_t)FS * D / 4);
    cvt_bf16_kernel<<<512,  256, 0, stream>>>((const float4*)sh_u,   (ushort4*)subb, (size_t)FS * D / 4);
    cvt_bf16_kernel<<<512,  256, 0, stream>>>((const float4*)sh_d,   (ushort4*)sdb,  (size_t)D * FS / 4);

    // -------- router + dispatch --------
    router_kernel<<<N, 64, 0, stream>>>(x, gate_w, topk_idx, topk_w, D);
    assign_kernel<<<(A + 255) / 256, 256, 0, stream>>>(topk_idx, topk_w, cnt,
                                                       slot_of_assign, token_of_slot,
                                                       weight_of_slot, A, C);
    dispatch_kernel<<<A, 256, 0, stream>>>(xb, slot_of_assign, buf, D);

    // -------- routed experts --------
    {   // [E] x [C,D] @ [F,D]^T -> silu*up -> hbuf [E,C,F]
        int tiles = (C / 32) * (F / 32);   // 48*44 = 2112
        gemm_swiglu_kernel<<<dim3((tiles + 7) / 8, E), 256, 0, stream>>>(
            buf, wgb, wub, hbuf, C, F, D,
            (long)C * D, (long)F * D, (long)C * F);
    }
    {   // [E] x [C,F] @ [D,F]^T -> weighted atomic scatter into y
        int tiles = (C / 32) * (D / 64);   // 48*32 = 1536
        gemm_down_scatter_kernel<<<dim3((tiles + 7) / 8, E), 256, 0, stream>>>(
            hbuf, wdb, token_of_slot, weight_of_slot, y, C, D, F);
    }

    // -------- shared expert --------
    {   // [N,D] @ [FS,D]^T -> silu*up -> hs [N,FS]
        int tiles = (N / 32) * (FS / 32);  // 256*88 = 22528
        gemm_swiglu_kernel<<<dim3((tiles + 7) / 8, 1), 256, 0, stream>>>(
            xb, sgb, subb, hs, N, FS, D, 0L, 0L, 0L);
    }
    {   // y += [N,FS] @ [D,FS]^T
        int tiles = (N / 32) * (D / 64);   // 256*32 = 8192
        gemm_down_add_kernel<<<dim3((tiles + 7) / 8, 1), 256, 0, stream>>>(
            hs, sdb, y, N, D, FS);
    }
}
